// GroupedMultiHeadAttention_59657095741920
// MI455X (gfx1250) — compile-verified, matrix operand order checked by reading, not run on hardware
//
#include <hip/hip_runtime.h>
#include <hip/hip_bf16.h>
#include <math.h>

#if defined(__has_builtin)
#if __has_builtin(__builtin_amdgcn_global_load_async_to_lds_b128)
#define HAVE_ASYNC_LDS 1
#endif
#endif

namespace {

constexpr int Bz = 4, Tt = 2048, Dd = 1024, Hh = 16, KVh = 4, DHd = 64, Gg = Hh / KVh;

typedef __attribute__((ext_vector_type(16))) __bf16 v16bf;
typedef __attribute__((ext_vector_type(8)))  float  v8f;
typedef __attribute__((ext_vector_type(4)))  int    v4i;

__device__ __forceinline__ v8f wmma_bf16(v16bf a, v16bf b, v8f c) {
  return __builtin_amdgcn_wmma_f32_16x16x32_bf16(false, a, false, b, (short)0, c,
                                                 false, false);
}

// 16-byte global -> LDS async copy (CDNA5 async engine, tracked by ASYNCcnt)
__device__ __forceinline__ void async_copy16(const void* g, void* l) {
#ifdef HAVE_ASYNC_LDS
  __builtin_amdgcn_global_load_async_to_lds_b128(
      (__attribute__((address_space(1))) v4i*)g,
      (__attribute__((address_space(3))) v4i*)l, 0, 0);
#else
  (void)g; (void)l;
#endif
}
__device__ __forceinline__ void wait_async0() {
#ifdef HAVE_ASYNC_LDS
#if __has_builtin(__builtin_amdgcn_s_wait_asynccnt)
  __builtin_amdgcn_s_wait_asynccnt(0);
#else
  asm volatile("s_wait_asynccnt 0" ::: "memory");
#endif
#endif
}

// DPP lane exchange (pure VALU; keeps the DS pipe free for WMMA operands)
template <int CTRL>
__device__ __forceinline__ float dpp_mov(float v) {
  return __int_as_float(
      __builtin_amdgcn_update_dpp(0, __float_as_int(v), CTRL, 0xf, 0xf, true));
}
__device__ __forceinline__ float redmax16(float v) {
  v = fmaxf(v, dpp_mov<0x0B1>(v));  // quad_perm(1,0,3,2)  (xor1)
  v = fmaxf(v, dpp_mov<0x04E>(v));  // quad_perm(2,3,0,1)  (xor2)
  v = fmaxf(v, dpp_mov<0x141>(v));  // row_half_mirror     (~xor4)
  v = fmaxf(v, dpp_mov<0x140>(v));  // row_mirror          (~xor8)
  return v;
}
__device__ __forceinline__ float redsum16(float v) {
  v += dpp_mov<0x0B1>(v);
  v += dpp_mov<0x04E>(v);
  v += dpp_mov<0x141>(v);
  v += dpp_mov<0x140>(v);
  return v;
}

// ---------------- fp32 -> bf16 convert ----------------
__global__ void f2bf_kernel(const float* __restrict__ in, __bf16* __restrict__ out, int n) {
  int i = blockIdx.x * blockDim.x + threadIdx.x;
  int stride = gridDim.x * blockDim.x;
  for (; i < n; i += stride) out[i] = (__bf16)in[i];
}

// ---------------- RoPE (in-place on bf16 q or k) ----------------
__global__ void rope_kernel(__bf16* __restrict__ q, int nh, int total) {
  int idx = blockIdx.x * blockDim.x + threadIdx.x;
  if (idx >= total) return;
  int i = idx & 31;
  int rest = idx >> 5;
  int h = rest % nh; rest /= nh;
  int t = rest % Tt;
  int b = rest / Tt;
  size_t base = (((size_t)b * Tt + t) * nh + h) * DHd;
  float inv_freq = __expf(-(float)(2 * i) * (9.210340371976184f / 64.0f));
  float sn, cs;
  __sincosf((float)t * inv_freq, &sn, &cs);
  float x1 = (float)q[base + i];
  float x2 = (float)q[base + 32 + i];
  q[base + i]      = (__bf16)(x1 * cs - x2 * sn);
  q[base + 32 + i] = (__bf16)(x1 * sn + x2 * cs);
}

// ---------------- tiled bf16 WMMA GEMM: C = A(MxK) @ B(KxN) ----------------
// Block tile 128x64, 8 waves, 2x2 WMMA tiles per wave. Double-buffered LDS:
// one barrier per k-step; A tile staged by the async engine, B by scatter.
template <bool OUT_BF16>
__global__ __launch_bounds__(256)
void gemm_bf16_kernel(const __bf16* __restrict__ A, const __bf16* __restrict__ Bw,
                      void* __restrict__ Cout, int M, int N, int K) {
  __shared__ __align__(16) __bf16 As[2][128][40];
  __shared__ __align__(16) __bf16 Bts[2][64][40];   // Bts[buf][n][k]
  const int tid  = threadIdx.x;
  const int lane = tid & 31;
  const int wave = tid >> 5;
  const int hi   = lane >> 4;
  const int ln   = lane & 15;
  const int mSub = wave & 3;
  const int nSub = wave >> 2;
  const int mBase = blockIdx.y * 128;
  const int nBase = blockIdx.x * 64;

  v8f acc[2][2];
  #pragma unroll
  for (int mi = 0; mi < 2; ++mi)
    #pragma unroll
    for (int ni = 0; ni < 2; ++ni)
      #pragma unroll
      for (int r = 0; r < 8; ++r) acc[mi][ni][r] = 0.0f;

  const int aRow = tid >> 1;
  const int aCol = (tid & 1) * 16;
  const int bK   = tid & 31;
  const int bN   = (tid >> 5) * 8;

  // stage k-step 0 into buffer 0
  {
    const __bf16* aSrc = A + (size_t)(mBase + aRow) * K + aCol;
#ifdef HAVE_ASYNC_LDS
    async_copy16(aSrc,     &As[0][aRow][aCol]);
    async_copy16(aSrc + 8, &As[0][aRow][aCol + 8]);
#else
    #pragma unroll
    for (int j = 0; j < 16; ++j) As[0][aRow][aCol + j] = aSrc[j];
#endif
    const __bf16* bSrc = Bw + (size_t)bK * N + nBase + bN;
    __bf16 breg[8];
    #pragma unroll
    for (int j = 0; j < 8; ++j) breg[j] = bSrc[j];
    #pragma unroll
    for (int j = 0; j < 8; ++j) Bts[0][bN + j][bK] = breg[j];
  }
  wait_async0();
  __syncthreads();

  int buf = 0;
  for (int kt = 0; kt < K; kt += 32) {
    const bool hasNext = (kt + 32 < K);
    if (hasNext) {  // stage next tile into buf^1 (no reader of buf^1 now)
      const __bf16* aSrc = A + (size_t)(mBase + aRow) * K + kt + 32 + aCol;
#ifdef HAVE_ASYNC_LDS
      async_copy16(aSrc,     &As[buf ^ 1][aRow][aCol]);
      async_copy16(aSrc + 8, &As[buf ^ 1][aRow][aCol + 8]);
#else
      __bf16 areg[16];
      #pragma unroll
      for (int j = 0; j < 16; ++j) areg[j] = aSrc[j];
      #pragma unroll
      for (int j = 0; j < 16; ++j) As[buf ^ 1][aRow][aCol + j] = areg[j];
#endif
      const __bf16* bSrc = Bw + (size_t)(kt + 32 + bK) * N + nBase + bN;
      __bf16 breg[8];
      #pragma unroll
      for (int j = 0; j < 8; ++j) breg[j] = bSrc[j];
      #pragma unroll
      for (int j = 0; j < 8; ++j) Bts[buf ^ 1][bN + j][bK] = breg[j];
    }

    // load all operands first, then issue the WMMA group
    v16bf av[2], bv[2];
    #pragma unroll
    for (int mi = 0; mi < 2; ++mi) {
      const __bf16* p = &As[buf][32 * mSub + 16 * mi + ln][8 * hi];
      #pragma unroll
      for (int j = 0; j < 8; ++j) { av[mi][j] = p[j]; av[mi][8 + j] = p[16 + j]; }
    }
    #pragma unroll
    for (int ni = 0; ni < 2; ++ni) {
      const __bf16* p = &Bts[buf][32 * nSub + 16 * ni + ln][8 * hi];
      #pragma unroll
      for (int j = 0; j < 8; ++j) { bv[ni][j] = p[j]; bv[ni][8 + j] = p[16 + j]; }
    }
    #pragma unroll
    for (int mi = 0; mi < 2; ++mi)
      #pragma unroll
      for (int ni = 0; ni < 2; ++ni)
        acc[mi][ni] = wmma_bf16(av[mi], bv[ni], acc[mi][ni]);

    if (hasNext) wait_async0();
    __syncthreads();
    buf ^= 1;
  }

  #pragma unroll
  for (int mi = 0; mi < 2; ++mi)
    #pragma unroll
    for (int ni = 0; ni < 2; ++ni)
      #pragma unroll
      for (int r = 0; r < 8; ++r) {
        int row = mBase + 32 * mSub + 16 * mi + 8 * hi + r;
        int col = nBase + 32 * nSub + 16 * ni + ln;
        float vv = acc[mi][ni][r];
        if (OUT_BF16) ((__bf16*)Cout)[(size_t)row * N + col] = (__bf16)vv;
        else          ((float*)Cout)[(size_t)row * N + col]  = vv;
      }
}

// ---------------- flash attention (GQA), one (b, head, 64-q-rows) per block ----------------
__global__ __launch_bounds__(128)
void attn_kernel(const __bf16* __restrict__ q, const __bf16* __restrict__ k,
                 const __bf16* __restrict__ v, const float* __restrict__ bias,
                 __bf16* __restrict__ y) {
  __shared__ __align__(16) __bf16 Kt[2][32][72];   // [buf][key][dh]
  __shared__ __align__(16) __bf16 Vt[2][64][40];   // [buf][dh][key] (transposed)
  __shared__ __align__(16) float  Sp[4][16][32];   // per-wave P scratch

  const int tid  = threadIdx.x;
  const int lane = tid & 31;
  const int wave = tid >> 5;
  const int hi   = lane >> 4;
  const int ln   = lane & 15;
  const int head = blockIdx.y;
  const int b    = blockIdx.z;
  const int kv   = head / Gg;
  const int qbase = blockIdx.x * 64 + wave * 16;
  const float scale = 0.125f;

  // Load this wave's 16 Q rows once (dh 0..31 and 32..63 A tiles)
  v16bf Qa[2];
  {
    const __bf16* qp = q + (((size_t)b * Tt + qbase + ln) * Hh + head) * DHd;
    #pragma unroll
    for (int s = 0; s < 2; ++s) {
      const __bf16* p = qp + 32 * s + 8 * hi;
      #pragma unroll
      for (int j = 0; j < 8; ++j) { Qa[s][j] = p[j]; Qa[s][8 + j] = p[16 + j]; }
    }
  }

  float m[8], l[8];
  v8f acc[4];
  #pragma unroll
  for (int r = 0; r < 8; ++r) { m[r] = -1.0e30f; l[r] = 0.0f; }
  #pragma unroll
  for (int t2 = 0; t2 < 4; ++t2)
    #pragma unroll
    for (int r = 0; r < 8; ++r) acc[t2][r] = 0.0f;

  const int ldRow = tid >> 2;           // key within tile
  const int ldCol = (tid & 3) * 16;     // dh chunk

  // stage tile 0 into buffer 0: K via async engine, V transposed by scatter
  {
    const __bf16* ks = k + (((size_t)b * Tt + ldRow) * KVh + kv) * DHd + ldCol;
    const __bf16* vs = v + (((size_t)b * Tt + ldRow) * KVh + kv) * DHd + ldCol;
#ifdef HAVE_ASYNC_LDS
    async_copy16(ks,     &Kt[0][ldRow][ldCol]);
    async_copy16(ks + 8, &Kt[0][ldRow][ldCol + 8]);
#else
    __bf16 kreg[16];
    #pragma unroll
    for (int j = 0; j < 16; ++j) kreg[j] = ks[j];
    #pragma unroll
    for (int j = 0; j < 16; ++j) Kt[0][ldRow][ldCol + j] = kreg[j];
#endif
    __bf16 vreg[16];
    #pragma unroll
    for (int j = 0; j < 16; ++j) vreg[j] = vs[j];
    #pragma unroll
    for (int j = 0; j < 16; ++j) Vt[0][ldCol + j][ldRow] = vreg[j];
  }
  wait_async0();
  __syncthreads();

  int buf = 0;
  for (int kb = 0; kb < Tt; kb += 32) {
    const bool hasNext = (kb + 32 < Tt);
    if (hasNext) {  // stage tile i+1 into buf^1 while computing tile i
      const __bf16* ks = k + (((size_t)b * Tt + kb + 32 + ldRow) * KVh + kv) * DHd + ldCol;
      const __bf16* vs = v + (((size_t)b * Tt + kb + 32 + ldRow) * KVh + kv) * DHd + ldCol;
#ifdef HAVE_ASYNC_LDS
      async_copy16(ks,     &Kt[buf ^ 1][ldRow][ldCol]);
      async_copy16(ks + 8, &Kt[buf ^ 1][ldRow][ldCol + 8]);
#else
      __bf16 kreg[16];
      #pragma unroll
      for (int j = 0; j < 16; ++j) kreg[j] = ks[j];
      #pragma unroll
      for (int j = 0; j < 16; ++j) Kt[buf ^ 1][ldRow][ldCol + j] = kreg[j];
#endif
      __bf16 vreg[16];
      #pragma unroll
      for (int j = 0; j < 16; ++j) vreg[j] = vs[j];
      #pragma unroll
      for (int j = 0; j < 16; ++j) Vt[buf ^ 1][ldCol + j][ldRow] = vreg[j];
    }

    // ---- S = Q @ K^T : load all 4 B operands first, then 4 WMMAs ----
    v16bf bk[2][2];
    #pragma unroll
    for (int sub = 0; sub < 2; ++sub)
      #pragma unroll
      for (int s = 0; s < 2; ++s) {
        const __bf16* p = &Kt[buf][16 * sub + ln][32 * s + 8 * hi];
        #pragma unroll
        for (int j = 0; j < 8; ++j) { bk[sub][s][j] = p[j]; bk[sub][s][8 + j] = p[16 + j]; }
      }
    v8f s0, s1;
    #pragma unroll
    for (int r = 0; r < 8; ++r) { s0[r] = 0.0f; s1[r] = 0.0f; }
    s0 = wmma_bf16(Qa[0], bk[0][0], s0);
    s1 = wmma_bf16(Qa[0], bk[1][0], s1);
    s0 = wmma_bf16(Qa[1], bk[0][1], s0);
    s1 = wmma_bf16(Qa[1], bk[1][1], s1);

    // ---- online softmax; P -> per-wave LDS scratch ----
    float* SpW = &Sp[wave][0][0];
    #pragma unroll
    for (int r = 0; r < 8; ++r) {
      int row = 8 * hi + r;
      int qt = qbase + row;
      const float* bp = bias + (size_t)qt * Tt + kb;
      float v0 = s0[r] * scale + bp[ln];
      float v1 = s1[r] * scale + bp[16 + ln];
      float mnew = fmaxf(m[r], redmax16(fmaxf(v0, v1)));
      float p0 = __expf(v0 - mnew);
      float p1 = __expf(v1 - mnew);
      float rs = redsum16(p0 + p1);
      float corr = __expf(m[r] - mnew);
      l[r] = l[r] * corr + rs;
      m[r] = mnew;
      #pragma unroll
      for (int t2 = 0; t2 < 4; ++t2) acc[t2][r] *= corr;
      SpW[row * 32 + ln]      = p0;
      SpW[row * 32 + 16 + ln] = p1;
    }
    // wave-private LDS RAW: DS pipe is in-order per wave, drain DScnt
    asm volatile("s_wait_dscnt 0" ::: "memory");

    // ---- O += P @ V : P as A operand, all 4 V operands loaded first ----
    v16bf Pa;
    {
      const float* pr = &SpW[ln * 32 + 8 * hi];
      #pragma unroll
      for (int j = 0; j < 8; ++j) { Pa[j] = (__bf16)pr[j]; Pa[8 + j] = (__bf16)pr[16 + j]; }
    }
    v16bf bvv[4];
    #pragma unroll
    for (int t2 = 0; t2 < 4; ++t2) {
      const __bf16* p = &Vt[buf][16 * t2 + ln][8 * hi];
      #pragma unroll
      for (int j = 0; j < 8; ++j) { bvv[t2][j] = p[j]; bvv[t2][8 + j] = p[16 + j]; }
    }
    #pragma unroll
    for (int t2 = 0; t2 < 4; ++t2) acc[t2] = wmma_bf16(Pa, bvv[t2], acc[t2]);

    if (hasNext) wait_async0();
    __syncthreads();
    buf ^= 1;
  }

  // normalize and write y (B,T,H*DH) as bf16 for the output projection
  #pragma unroll
  for (int t2 = 0; t2 < 4; ++t2)
    #pragma unroll
    for (int r = 0; r < 8; ++r) {
      int row = 8 * hi + r;
      int qt = qbase + row;
      float ov = acc[t2][r] / l[r];
      y[(((size_t)b * Tt + qt) * Hh + head) * DHd + 16 * t2 + ln] = (__bf16)ov;
    }
}

} // namespace

extern "C" void kernel_launch(void* const* d_in, const int* in_sizes, int n_in,
                              void* d_out, int out_size, void* d_ws, size_t ws_size,
                              hipStream_t stream) {
  (void)in_sizes; (void)n_in; (void)out_size; (void)ws_size;
  const float* x    = (const float*)d_in[0];
  const float* bias = (const float*)d_in[1];
  const float* Wq   = (const float*)d_in[2];
  const float* Wk   = (const float*)d_in[3];
  const float* Wv   = (const float*)d_in[4];
  const float* Wo   = (const float*)d_in[5];

  char* ws = (char*)d_ws;
  size_t off = 0;
  auto carve = [&](size_t elems) {
    void* p = ws + off;
    off += (elems * sizeof(__bf16) + 255) & ~(size_t)255;
    return (__bf16*)p;
  };
  __bf16* xbf = carve((size_t)Bz * Tt * Dd);
  __bf16* wqb = carve((size_t)Dd * Hh * DHd);
  __bf16* wkb = carve((size_t)Dd * KVh * DHd);
  __bf16* wvb = carve((size_t)Dd * KVh * DHd);
  __bf16* wob = carve((size_t)Hh * DHd * Dd);
  __bf16* qbf = carve((size_t)Bz * Tt * Hh * DHd);
  __bf16* kbf = carve((size_t)Bz * Tt * KVh * DHd);
  __bf16* vbf = carve((size_t)Bz * Tt * KVh * DHd);
  __bf16* ybf = carve((size_t)Bz * Tt * Hh * DHd);

  auto cvt = [&](const float* src, __bf16* dst, int n) {
    int blocks = (n + 255) / 256;
    if (blocks > 8192) blocks = 8192;
    f2bf_kernel<<<blocks, 256, 0, stream>>>(src, dst, n);
  };
  cvt(x,  xbf, Bz * Tt * Dd);
  cvt(Wq, wqb, Dd * Hh * DHd);
  cvt(Wk, wkb, Dd * KVh * DHd);
  cvt(Wv, wvb, Dd * KVh * DHd);
  cvt(Wo, wob, Hh * DHd * Dd);

  const int M = Bz * Tt;                 // 8192
  dim3 gq((Hh * DHd) / 64, M / 128);     // (16, 64)
  dim3 gk((KVh * DHd) / 64, M / 128);    // (4, 64)
  gemm_bf16_kernel<true><<<gq, 256, 0, stream>>>(xbf, wqb, qbf, M, Hh * DHd, Dd);
  gemm_bf16_kernel<true><<<gk, 256, 0, stream>>>(xbf, wkb, kbf, M, KVh * DHd, Dd);
  gemm_bf16_kernel<true><<<gk, 256, 0, stream>>>(xbf, wvb, vbf, M, KVh * DHd, Dd);

  rope_kernel<<<(Bz * Tt * Hh * 32) / 256, 256, 0, stream>>>(qbf, Hh, Bz * Tt * Hh * 32);
  rope_kernel<<<(Bz * Tt * KVh * 32) / 256, 256, 0, stream>>>(kbf, KVh, Bz * Tt * KVh * 32);

  dim3 ga(Tt / 64, Hh, Bz);              // (32, 16, 4)
  attn_kernel<<<ga, 128, 0, stream>>>(qbf, kbf, vbf, bias, ybf);

  gemm_bf16_kernel<false><<<gq, 256, 0, stream>>>(ybf, wob, d_out, M, Dd, Hh * DHd);
}
